// LocalTransformer_80513456931527
// MI455X (gfx1250) — matrix operand order repaired and checked by reading, not compiled, not run on hardware
//
#include <hip/hip_runtime.h>
#include <math.h>

// ---------------- problem constants ----------------
#define B_SZ   2
#define N_PTS  8192
#define TOKENS (B_SZ * N_PTS)   // 16384
#define CH     64               // feature / hidden width
#define KNN    16               // neighbors

typedef __attribute__((ext_vector_type(16))) __bf16 bf16x16;
typedef __attribute__((ext_vector_type(8)))  __bf16 bf16x8;
typedef __attribute__((ext_vector_type(8)))  float  f32x8;

// ---------------- WMMA fragment helpers ----------------
__device__ __forceinline__ bf16x16 mk16(bf16x8 lo, bf16x8 hi) {
    bf16x16 r;
#pragma unroll
    for (int i = 0; i < 8; ++i) { r[i] = lo[i]; r[i + 8] = hi[i]; }
    return r;
}

// A-matrix (16x32 bf16) fragment from a row-major 16x64 bf16 staging buffer.
// Lane L holds row (L&15); half-slots 0-7 = K base..base+7, 8-15 = K base+16..+23,
// base = (L<16 ? 0 : 8) within the 32-wide K chunk `ch`.
__device__ __forceinline__ bf16x16 loadA(const __bf16* stage, int lane, int chunk) {
    int row  = lane & 15;
    int base = (lane < 16) ? 0 : 8;
    int k0   = chunk * 32 + base;
    bf16x8 lo = *(const bf16x8*)(stage + row * CH + k0);
    bf16x8 hi = *(const bf16x8*)(stage + row * CH + k0 + 16);
    return mk16(lo, hi);
}

// B-matrix (32x16 bf16) fragment from an n-major transposed weight W^T[n*64+k].
// Lane L holds column n = t*16 + (L&15); half-slots 0-7 = K base..base+7,
// 8-15 = K base+16..+23, base = chunk*32 + (L<16 ? 0 : 8).
__device__ __forceinline__ bf16x16 loadB(const __bf16* WT, int lane, int t, int chunk) {
    int n  = t * 16 + (lane & 15);
    int kb = chunk * 32 + ((lane < 16) ? 0 : 8);
    const bf16x8* p = (const bf16x8*)(WT + n * CH + kb);
    return mk16(p[0], p[2]);
}

__device__ __forceinline__ f32x8 wmma_bf16(bf16x16 a, bf16x16 b, f32x8 c) {
    return __builtin_amdgcn_wmma_f32_16x16x32_bf16(false, a, false, b,
                                                   (short)0, c, false, false);
}

// ---------------- Kernel 1: KNN (f32 VALU, top-16 insertion in registers) ----
__global__ void __launch_bounds__(256)
knn_kernel(const float* __restrict__ xyzp, int* __restrict__ idx_out) {
    __shared__ float cx[256][4];
    const int tid = threadIdx.x;
    const int b   = blockIdx.x / (N_PTS / 256);
    const int qi  = (blockIdx.x % (N_PTS / 256)) * 256 + tid;
    const int gq  = b * N_PTS + qi;

    const float qx = xyzp[gq * 4 + 0];
    const float qy = xyzp[gq * 4 + 1];
    const float qz = xyzp[gq * 4 + 2];

    float bd[KNN]; int bi[KNN];
#pragma unroll
    for (int s = 0; s < KNN; ++s) { bd[s] = 3.0e38f; bi[s] = b * N_PTS; }

    for (int c0 = 0; c0 < N_PTS; c0 += 256) {
        __syncthreads();
        float4 v = *(const float4*)(xyzp + (size_t)(b * N_PTS + c0 + tid) * 4);
        cx[tid][0] = v.x; cx[tid][1] = v.y; cx[tid][2] = v.z;
        __syncthreads();
#pragma unroll 4
        for (int j = 0; j < 256; ++j) {
            float dx = qx - cx[j][0];
            float dy = qy - cx[j][1];
            float dz = qz - cx[j][2];
            float d  = dx * dx + dy * dy + dz * dz;
            if (d < bd[KNN - 1]) {
                float cd = d; int ci = b * N_PTS + c0 + j;   // store globalized idx
#pragma unroll
                for (int s = 0; s < KNN; ++s) {
                    if (cd < bd[s]) {
                        float td = bd[s]; int ti = bi[s];
                        bd[s] = cd; bi[s] = ci; cd = td; ci = ti;
                    }
                }
            }
        }
    }
#pragma unroll
    for (int s = 0; s < KNN; ++s) idx_out[(size_t)gq * KNN + s] = bi[s];
}

// ---------------- Kernel 2: weight fusion + bf16 transposes --------------
// Wq' = Wk@Wq, Wks' = Wk@Wks, Wv' = Wk@Wv   (x = f@Wk+bk is linear)
// Wpt = Wp2@Wt1, bpt = bp2@Wt1 + bt1        (pos_enc contribution to a_in@Wt1)
__global__ void __launch_bounds__(256)
fuse_kernel(const float* __restrict__ Wk,  const float* __restrict__ bk,
            const float* __restrict__ Wq,  const float* __restrict__ Wks,
            const float* __restrict__ Wv,  const float* __restrict__ Wp2,
            const float* __restrict__ bp2, const float* __restrict__ Wt1,
            const float* __restrict__ bt1, const float* __restrict__ Wt2,
            __bf16* WqT, __bf16* WksT, __bf16* WvT,
            __bf16* Wp2T, __bf16* WptT, __bf16* Wt1T, __bf16* Wt2T,
            float* bq, float* bks, float* bv, float* bpt) {
    const int tid = threadIdx.x;
#pragma unroll 1
    for (int i = 0; i < 16; ++i) {
        int e = tid * 16 + i;
        int k = e >> 6, n = e & 63;
        float s1 = 0.f, s2 = 0.f, s3 = 0.f, s4 = 0.f;
        for (int j = 0; j < CH; ++j) {
            float a = Wk[k * CH + j];
            s1 += a * Wq[j * CH + n];
            s2 += a * Wks[j * CH + n];
            s3 += a * Wv[j * CH + n];
            s4 += Wp2[k * CH + j] * Wt1[j * CH + n];
        }
        WqT [n * CH + k] = (__bf16)s1;
        WksT[n * CH + k] = (__bf16)s2;
        WvT [n * CH + k] = (__bf16)s3;
        WptT[n * CH + k] = (__bf16)s4;
        Wp2T[n * CH + k] = (__bf16)Wp2[k * CH + n];
        Wt1T[n * CH + k] = (__bf16)Wt1[k * CH + n];
        Wt2T[n * CH + k] = (__bf16)Wt2[k * CH + n];
    }
    if (tid < CH) {
        int n = tid;
        float s1 = 0.f, s2 = 0.f, s3 = 0.f, s4 = 0.f;
        for (int j = 0; j < CH; ++j) {
            float a = bk[j];
            s1 += a * Wq[j * CH + n];
            s2 += a * Wks[j * CH + n];
            s3 += a * Wv[j * CH + n];
            s4 += bp2[j] * Wt1[j * CH + n];
        }
        bq[n] = s1; bks[n] = s2; bv[n] = s3; bpt[n] = s4 + bt1[n];
    }
}

// ---------------- Kernel 3: fused q/k/v projections (WMMA) ----------------
__global__ void __launch_bounds__(256)
qkv_kernel(const float* __restrict__ feat,
           const __bf16* __restrict__ WqT, const __bf16* __restrict__ WksT,
           const __bf16* __restrict__ WvT,
           const float* __restrict__ bq, const float* __restrict__ bks,
           const float* __restrict__ bv,
           float* __restrict__ q, float* __restrict__ ks, float* __restrict__ vv) {
    const int lane = threadIdx.x & 31;
    const int wave = threadIdx.x >> 5;
    const int rb   = blockIdx.x * 128 + wave * 16;   // 16 token rows per wave
    const int row  = rb + (lane & 15);
    const int base = (lane < 16) ? 0 : 8;
    const int hf   = lane >> 4;

    bf16x16 a[2];
#pragma unroll
    for (int c = 0; c < 2; ++c) {
        int k0 = c * 32 + base;
        const float* fp = feat + (size_t)row * CH + k0;
        float4 f0 = *(const float4*)(fp);
        float4 f1 = *(const float4*)(fp + 4);
        float4 f2 = *(const float4*)(fp + 16);
        float4 f3 = *(const float4*)(fp + 20);
        bf16x16 t;
        t[0] = (__bf16)f0.x; t[1] = (__bf16)f0.y; t[2]  = (__bf16)f0.z; t[3]  = (__bf16)f0.w;
        t[4] = (__bf16)f1.x; t[5] = (__bf16)f1.y; t[6]  = (__bf16)f1.z; t[7]  = (__bf16)f1.w;
        t[8] = (__bf16)f2.x; t[9] = (__bf16)f2.y; t[10] = (__bf16)f2.z; t[11] = (__bf16)f2.w;
        t[12] = (__bf16)f3.x; t[13] = (__bf16)f3.y; t[14] = (__bf16)f3.z; t[15] = (__bf16)f3.w;
        a[c] = t;
    }

    auto run = [&](const __bf16* WT, const float* bias, float* outp) {
#pragma unroll
        for (int t = 0; t < 4; ++t) {
            int col = t * 16 + (lane & 15);
            float bb = bias[col];
            f32x8 acc;
#pragma unroll
            for (int r = 0; r < 8; ++r) acc[r] = bb;
            acc = wmma_bf16(a[0], loadB(WT, lane, t, 0), acc);
            acc = wmma_bf16(a[1], loadB(WT, lane, t, 1), acc);
#pragma unroll
            for (int r = 0; r < 8; ++r)
                outp[(size_t)(rb + r + 8 * hf) * CH + col] = acc[r];
        }
    };
    run(WqT,  bq,  q);
    run(WksT, bks, ks);
    run(WvT,  bv,  vv);
}

// ---------------- Kernel 4: local attention (WMMA, one wave per token) ----
#define WAVES_C 8
__global__ void __launch_bounds__(256)
attn_kernel(const float* __restrict__ xyzp, const float* __restrict__ feat,
            const float* __restrict__ Wp1,  const float* __restrict__ bp1,
            const float* __restrict__ bp2,  const float* __restrict__ bt2,
            const float* __restrict__ Wa,   const float* __restrict__ ba,
            const int* __restrict__ idx,
            const float* __restrict__ qg,  const float* __restrict__ ksg,
            const float* __restrict__ vvg,
            const __bf16* __restrict__ Wp2T, const __bf16* __restrict__ WptT,
            const __bf16* __restrict__ Wt1T, const __bf16* __restrict__ Wt2T,
            const float* __restrict__ bpt,
            float* __restrict__ out) {
    __shared__ __bf16 bufP[WAVES_C][16 * CH];   // pos1, later reused for h
    __shared__ __bf16 bufQ[WAVES_C][16 * CH];   // q - k
    __shared__ float  sres[WAVES_C][CH];
    __shared__ float  sWp1[4 * CH];
    __shared__ float  sbp1[CH];

    const int tid  = threadIdx.x;
    const int lane = tid & 31;
    const int w    = tid >> 5;
    const int token = blockIdx.x * WAVES_C + w;   // flattened b*N + n
    const int r    = lane & 15;                   // neighbor row this lane stages
    const int hf   = lane >> 4;
    const int ch2  = hf * 32;                     // column half this lane stages

    sWp1[tid] = Wp1[tid];                         // 256 threads, 256 entries
    if (tid < CH) sbp1[tid] = bp1[tid];

    const int gid = idx[(size_t)token * KNN + r];
    const float rel0 = xyzp[token * 4 + 0] - xyzp[gid * 4 + 0];
    const float rel1 = xyzp[token * 4 + 1] - xyzp[gid * 4 + 1];
    const float rel2 = xyzp[token * 4 + 2] - xyzp[gid * 4 + 2];
    const float rel3 = xyzp[token * 4 + 3] - xyzp[gid * 4 + 3];

    __syncthreads();

    // stage pos1 = relu(rel @ Wp1 + bp1) and (q - k), bf16 row-major
#pragma unroll 4
    for (int c = 0; c < 32; ++c) {
        int cc = ch2 + c;
        float p = sbp1[cc] + rel0 * sWp1[cc] + rel1 * sWp1[CH + cc]
                           + rel2 * sWp1[2 * CH + cc] + rel3 * sWp1[3 * CH + cc];
        bufP[w][r * CH + cc] = (__bf16)fmaxf(p, 0.f);
    }
#pragma unroll
    for (int c4 = 0; c4 < 32; c4 += 4) {
        float4 qv = *(const float4*)(qg  + (size_t)token * CH + ch2 + c4);
        float4 kv = *(const float4*)(ksg + (size_t)gid   * CH + ch2 + c4);
        bufQ[w][r * CH + ch2 + c4 + 0] = (__bf16)(qv.x - kv.x);
        bufQ[w][r * CH + ch2 + c4 + 1] = (__bf16)(qv.y - kv.y);
        bufQ[w][r * CH + ch2 + c4 + 2] = (__bf16)(qv.z - kv.z);
        bufQ[w][r * CH + ch2 + c4 + 3] = (__bf16)(qv.w - kv.w);
    }
    __syncthreads();

    bf16x16 aP0 = loadA(bufP[w], lane, 0), aP1 = loadA(bufP[w], lane, 1);
    bf16x16 aQ0 = loadA(bufQ[w], lane, 0), aQ1 = loadA(bufQ[w], lane, 1);

    f32x8 pe[4];
#pragma unroll
    for (int t = 0; t < 4; ++t) {
        int col = t * 16 + r;
        // pos_enc tile = pos1 @ Wp2 + bp2   (kept for (v + pos_enc))
        f32x8 acc;
        float bb = bp2[col];
#pragma unroll
        for (int rr = 0; rr < 8; ++rr) acc[rr] = bb;
        acc = wmma_bf16(aP0, loadB(Wp2T, lane, t, 0), acc);
        acc = wmma_bf16(aP1, loadB(Wp2T, lane, t, 1), acc);
        pe[t] = acc;
        // a_in @ Wt1 = pos1 @ (Wp2 Wt1) + (q-k) @ Wt1 + bpt
        f32x8 h;
        float bh = bpt[col];
#pragma unroll
        for (int rr = 0; rr < 8; ++rr) h[rr] = bh;
        h = wmma_bf16(aP0, loadB(WptT, lane, t, 0), h);
        h = wmma_bf16(aP1, loadB(WptT, lane, t, 1), h);
        h = wmma_bf16(aQ0, loadB(Wt1T, lane, t, 0), h);
        h = wmma_bf16(aQ1, loadB(Wt1T, lane, t, 1), h);
        // relu, restage into bufP (aP already consumed into registers)
#pragma unroll
        for (int rr = 0; rr < 8; ++rr)
            bufP[w][(rr + 8 * hf) * CH + col] = (__bf16)fmaxf(h[rr], 0.f);
    }
    __syncthreads();

    bf16x16 aH0 = loadA(bufP[w], lane, 0), aH1 = loadA(bufP[w], lane, 1);

    int gi[8];
#pragma unroll
    for (int rr = 0; rr < 8; ++rr)
        gi[rr] = idx[(size_t)token * KNN + rr + 8 * hf];

    float res[4];
#pragma unroll
    for (int t = 0; t < 4; ++t) {
        int col = t * 16 + r;
        f32x8 at;
        float bb = bt2[col];
#pragma unroll
        for (int rr = 0; rr < 8; ++rr) at[rr] = bb;
        at = wmma_bf16(aH0, loadB(Wt2T, lane, t, 0), at);
        at = wmma_bf16(aH1, loadB(Wt2T, lane, t, 1), at);
        // softmax over the 16 neighbors (M axis): 8 rows here + lane-half swap
        float s[8];
        float mx = -3.0e38f;
#pragma unroll
        for (int rr = 0; rr < 8; ++rr) { s[rr] = at[rr] * 0.125f; mx = fmaxf(mx, s[rr]); }
        mx = fmaxf(mx, __shfl_xor(mx, 16, 32));
        float ev[8];
        float sum = 0.f;
#pragma unroll
        for (int rr = 0; rr < 8; ++rr) { ev[rr] = __expf(s[rr] - mx); sum += ev[rr]; }
        sum += __shfl_xor(sum, 16, 32);
        float inv = 1.0f / sum;
        // res = sum_k attn * (v + pos_enc)
        float acc = 0.f;
#pragma unroll
        for (int rr = 0; rr < 8; ++rr) {
            float vval = vvg[(size_t)gi[rr] * CH + col] + pe[t][rr];
            acc += ev[rr] * inv * vval;
        }
        acc += __shfl_xor(acc, 16, 32);
        res[t] = acc;
    }
    if (hf == 0) {
#pragma unroll
        for (int t = 0; t < 4; ++t) sres[w][t * 16 + r] = res[t];
    }
    __syncthreads();

    // out = res @ Wa + ba + features : each lane produces 2 channels
#pragma unroll
    for (int o = 0; o < 2; ++o) {
        int c = lane * 2 + o;
        float s = ba[c] + feat[(size_t)token * CH + c];
        for (int k = 0; k < CH; ++k) s += sres[w][k] * Wa[k * CH + c];
        out[(size_t)token * CH + c] = s;
    }
}

// ---------------- launcher ----------------
extern "C" void kernel_launch(void* const* d_in, const int* in_sizes, int n_in,
                              void* d_out, int out_size, void* d_ws, size_t ws_size,
                              hipStream_t stream) {
    (void)in_sizes; (void)n_in; (void)out_size; (void)ws_size;
    const float* xyzp = (const float*)d_in[0];
    const float* feat = (const float*)d_in[1];
    const float* Wk   = (const float*)d_in[2];
    const float* bk   = (const float*)d_in[3];
    const float* Wq   = (const float*)d_in[4];
    const float* Wks  = (const float*)d_in[5];
    const float* Wv   = (const float*)d_in[6];
    const float* Wp1  = (const float*)d_in[7];
    const float* bp1  = (const float*)d_in[8];
    const float* Wp2  = (const float*)d_in[9];
    const float* bp2  = (const float*)d_in[10];
    const float* Wt1  = (const float*)d_in[11];
    const float* bt1  = (const float*)d_in[12];
    const float* Wt2  = (const float*)d_in[13];
    const float* bt2  = (const float*)d_in[14];
    const float* Wa   = (const float*)d_in[15];
    const float* ba   = (const float*)d_in[16];
    float* out = (float*)d_out;

    char* ws = (char*)d_ws;
    const size_t OFF_IDX = 0;                                  // 16384*16*4 = 1 MiB
    const size_t OFF_Q   = (size_t)TOKENS * KNN * 4;
    const size_t OFF_K   = OFF_Q + (size_t)TOKENS * CH * 4;
    const size_t OFF_V   = OFF_K + (size_t)TOKENS * CH * 4;
    const size_t OFF_WQ  = OFF_V + (size_t)TOKENS * CH * 4;
    const size_t OFF_WKS = OFF_WQ  + CH * CH * 2;
    const size_t OFF_WV  = OFF_WKS + CH * CH * 2;
    const size_t OFF_WP2 = OFF_WV  + CH * CH * 2;
    const size_t OFF_WPT = OFF_WP2 + CH * CH * 2;
    const size_t OFF_WT1 = OFF_WPT + CH * CH * 2;
    const size_t OFF_WT2 = OFF_WT1 + CH * CH * 2;
    const size_t OFF_BQ  = OFF_WT2 + CH * CH * 2;
    const size_t OFF_BKS = OFF_BQ  + CH * 4;
    const size_t OFF_BV  = OFF_BKS + CH * 4;
    const size_t OFF_BPT = OFF_BV  + CH * 4;

    int*    idxb = (int*)(ws + OFF_IDX);
    float*  qb   = (float*)(ws + OFF_Q);
    float*  kb   = (float*)(ws + OFF_K);
    float*  vb   = (float*)(ws + OFF_V);
    __bf16* WqT  = (__bf16*)(ws + OFF_WQ);
    __bf16* WksT = (__bf16*)(ws + OFF_WKS);
    __bf16* WvT  = (__bf16*)(ws + OFF_WV);
    __bf16* Wp2T = (__bf16*)(ws + OFF_WP2);
    __bf16* WptT = (__bf16*)(ws + OFF_WPT);
    __bf16* Wt1T = (__bf16*)(ws + OFF_WT1);
    __bf16* Wt2T = (__bf16*)(ws + OFF_WT2);
    float*  bqf  = (float*)(ws + OFF_BQ);
    float*  bksf = (float*)(ws + OFF_BKS);
    float*  bvf  = (float*)(ws + OFF_BV);
    float*  bptf = (float*)(ws + OFF_BPT);

    knn_kernel<<<TOKENS / 256, 256, 0, stream>>>(xyzp, idxb);
    fuse_kernel<<<1, 256, 0, stream>>>(Wk, bk, Wq, Wks, Wv, Wp2, bp2, Wt1, bt1, Wt2,
                                       WqT, WksT, WvT, Wp2T, WptT, Wt1T, Wt2T,
                                       bqf, bksf, bvf, bptf);
    qkv_kernel<<<TOKENS / 128, 256, 0, stream>>>(feat, WqT, WksT, WvT,
                                                 bqf, bksf, bvf, qb, kb, vb);
    attn_kernel<<<TOKENS / WAVES_C, 256, 0, stream>>>(xyzp, feat, Wp1, bp1, bp2, bt2,
                                                      Wa, ba, idxb, qb, kb, vb,
                                                      Wp2T, WptT, Wt1T, Wt2T, bptf, out);
}
